// MACoreAttention_67353677136410
// MI455X (gfx1250) — matrix operand order repaired and checked by reading, not compiled, not run on hardware
//
#include <hip/hip_runtime.h>
#include <math.h>

// ---------------------------------------------------------------------------
// Sliding-window attention (flash-style, online softmax), CDNA5 / gfx1250.
// B=2, S=2048, H=8, D=64, window radius W read from device memory.
// One wave32 per 16-query tile; bf16 WMMA 16x16x32 for both QK^T and PV.
// ---------------------------------------------------------------------------

typedef __attribute__((ext_vector_type(16))) __bf16 v16bf;
typedef __attribute__((ext_vector_type(8)))  float  v8f;
typedef __attribute__((ext_vector_type(4)))  float  f4;

constexpr int BB = 2;
constexpr int SS = 2048;
constexpr int HH = 8;
constexpr int DD = 64;
constexpr int QT = 16;              // queries per tile (WMMA M)
constexpr int KT = 32;              // keys per main-loop iteration
constexpr int RS = HH * DD;         // float stride between consecutive seq rows

__device__ __forceinline__ float xor16(float x) { return __shfl_xor(x, 16, 32); }

// Load one row of a 16x32 bf16 A-operand in contiguous k order:
// element e -> d = dbase + 16*hf + e   (one contiguous 16-float run)
__device__ __forceinline__ v16bf load_A_row(const float* __restrict__ row,
                                            int dbase, int hf) {
  v16bf a;
  const float* p0 = row + dbase + 16 * hf;
  const float* p1 = row + dbase + 16 * hf + 8;
#pragma unroll
  for (int e = 0; e < 8; ++e) {
    a[e]     = (__bf16)p0[e];
    a[8 + e] = (__bf16)p1[e];
  }
  return a;
}

__global__ __launch_bounds__(32)
void swa_attn_fwd_gfx1250(const float* __restrict__ Q,
                          const float* __restrict__ K,
                          const float* __restrict__ V,
                          const int*   __restrict__ Wp,
                          float*       __restrict__ O) {
  const int Wv   = Wp[0];
  const int ntq  = SS / QT;                 // 128 query tiles per (b,h)
  const int tile = blockIdx.x;
  const int qt   = tile % ntq;
  const int bh   = tile / ntq;
  const int h    = bh % HH;
  const int b    = bh / HH;
  const int i0   = qt * QT;

  const int lane = threadIdx.x;             // 0..31
  const int qn   = lane & 15;               // query column (matmul-1), D column (matmul-2)
  const int hf   = lane >> 4;               // wave half

  const float* Qb = Q + ((size_t)(b * SS) * HH + h) * DD;
  const float* Kb = K + ((size_t)(b * SS) * HH + h) * DD;
  const float* Vb = V + ((size_t)(b * SS) * HH + h) * DD;
  float*       Ob = O + ((size_t)(b * SS) * HH + h) * DD;

  const float scale = 0.125f;               // 1/sqrt(64)

  // --- Q as B-operand of S^T = K * Q^T. B layout: element e -> d = 32c + 16*hf + e
  v16bf qb[2];
  {
    const float* qrow = Qb + (size_t)(i0 + qn) * RS;
#pragma unroll
    for (int c = 0; c < 2; ++c) {
      const int d0 = 32 * c + 16 * hf;
#pragma unroll
      for (int g = 0; g < 4; ++g) {
        f4 v = *(const f4*)(qrow + d0 + 4 * g);
#pragma unroll
        for (int r = 0; r < 4; ++r) qb[c][4 * g + r] = (__bf16)(v[r] * scale);
      }
    }
  }

  // Online-softmax state (per query q = lane&15, duplicated across halves)
  float m_i = -1e30f;
  float l_i = 0.0f;
  v8f o[4];
#pragma unroll
  for (int c = 0; c < 4; ++c)
#pragma unroll
    for (int v = 0; v < 8; ++v) o[c][v] = 0.0f;

  const int iq = i0 + qn;                   // this lane's query index
  int jb_begin = i0 - Wv; if (jb_begin < 0) jb_begin = 0;
  jb_begin &= ~15;
  int jb_end = i0 + QT + Wv; if (jb_end > SS) jb_end = SS;

  const v8f vzero = {0.f, 0.f, 0.f, 0.f, 0.f, 0.f, 0.f, 0.f};

  for (int jb = jb_begin; jb < jb_end; jb += KT) {
    // ---- S^T = K * Q^T : two 16-key blocks, each chained over D=64 (2 x K=32)
    v8f s[2];
#pragma unroll
    for (int blk = 0; blk < 2; ++blk) {
      int kr = jb + 16 * blk + qn;
      if (kr > SS - 1) kr = SS - 1;         // clamped load, masked below
      const float* krow = Kb + (size_t)kr * RS;
      v16bf a0 = load_A_row(krow, 0, hf);
      v16bf a1 = load_A_row(krow, 32, hf);
      v8f acc = vzero;
      acc = __builtin_amdgcn_wmma_f32_16x16x32_bf16(false, a0, false, qb[0],
                                                    (short)0, acc, false, false);
      acc = __builtin_amdgcn_wmma_f32_16x16x32_bf16(false, a1, false, qb[1],
                                                    (short)0, acc, false, false);
      s[blk] = acc;
    }

    // ---- band mask + block max (row stats per q = lane&15; rows = keys)
    float bm = -1e30f;
#pragma unroll
    for (int blk = 0; blk < 2; ++blk) {
#pragma unroll
      for (int v = 0; v < 8; ++v) {
        int j = jb + 16 * blk + v + 8 * hf;
        int dlt = iq - j; if (dlt < 0) dlt = -dlt;
        bool ok = (j < SS) && (dlt <= Wv);
        float sv = ok ? s[blk][v] : -1e30f;
        s[blk][v] = sv;
        bm = fmaxf(bm, sv);
      }
    }
    bm = fmaxf(bm, xor16(bm));
    const float m_new   = fmaxf(m_i, bm);
    const float rescale = __expf(m_i - m_new);   // 1.0 when both are -1e30

    // ---- exponentiate, row sums
    float rs = 0.0f;
#pragma unroll
    for (int blk = 0; blk < 2; ++blk)
#pragma unroll
      for (int v = 0; v < 8; ++v) {
        float p = __expf(s[blk][v] - m_new);
        s[blk][v] = p;
        rs += p;
      }
    rs += xor16(rs);
    l_i = l_i * rescale + rs;
    m_i = m_new;

    // ---- rescale O accumulators; row q' = v + 8*hf, factor lives on lane q'
#pragma unroll
    for (int v = 0; v < 8; ++v) {
      float rsc = __shfl(rescale, v + 8 * hf, 32);
#pragma unroll
      for (int c = 0; c < 4; ++c) o[c][v] *= rsc;
    }

    // ---- P as A-operand (no data movement: D layout == A layout here)
    //      element e<8  -> key 8*hf + e        == s[0][e]
    //      element e>=8 -> key 16 + 8*hf + e-8 == s[1][e-8]
    v16bf pa;
#pragma unroll
    for (int e = 0; e < 8; ++e) {
      pa[e]     = (__bf16)s[0][e];
      pa[8 + e] = (__bf16)s[1][e];
    }

    // ---- O += P * V : 4 D-chunks of 16; B layout: element e -> key jb + 8*hf + (e&7) + 16*(e>>3)
#pragma unroll
    for (int c = 0; c < 4; ++c) {
      v16bf vb;
#pragma unroll
      for (int e = 0; e < 16; ++e) {
        int vr = jb + 8 * hf + (e & 7) + 16 * (e >> 3);
        if (vr > SS - 1) vr = SS - 1;       // clamped; P==0 there after flush
        vb[e] = (__bf16)Vb[(size_t)vr * RS + 16 * c + qn];
      }
      o[c] = __builtin_amdgcn_wmma_f32_16x16x32_bf16(false, pa, false, vb,
                                                     (short)0, o[c], false, false);
    }
  }

  // ---- finalize: divide by row sum, write out (lane spans contiguous D)
#pragma unroll
  for (int v = 0; v < 8; ++v) {
    float lv  = __shfl(l_i, v + 8 * hf, 32);
    float inv = 1.0f / lv;
    const int qrow = i0 + v + 8 * hf;
    float* orow = Ob + (size_t)qrow * RS;
#pragma unroll
    for (int c = 0; c < 4; ++c) orow[16 * c + qn] = o[c][v] * inv;
  }
}

extern "C" void kernel_launch(void* const* d_in, const int* in_sizes, int n_in,
                              void* d_out, int out_size, void* d_ws, size_t ws_size,
                              hipStream_t stream) {
  (void)in_sizes; (void)n_in; (void)out_size; (void)d_ws; (void)ws_size;
  const float* Q  = (const float*)d_in[0];
  const float* K  = (const float*)d_in[1];
  const float* V  = (const float*)d_in[2];
  const int*   Wp = (const int*)d_in[3];
  float* Out = (float*)d_out;

  const int nblocks = BB * HH * (SS / QT);   // 2048 wave-tiles
  swa_attn_fwd_gfx1250<<<nblocks, 32, 0, stream>>>(Q, K, V, Wp, Out);
}
